// RNN_72361609003238
// MI455X (gfx1250) — matrix operand order repaired and checked by reading, not compiled, not run on hardware
//
#include <hip/hip_runtime.h>

#define SEQ   512
#define BATCH 256
#define INSZ  64
#define HID   256
#define HROW  264   // padded hbuf row (bf16): 528B = 33*16B -> keeps 16B align, rotates LDS banks

typedef __bf16 bf16_t;
typedef __attribute__((ext_vector_type(16))) __bf16 v16bf;
typedef __attribute__((ext_vector_type(8)))  __bf16 v8bf;
typedef __attribute__((ext_vector_type(8)))  float  v8f;

#if __has_builtin(__builtin_amdgcn_tanhf)
__device__ __forceinline__ float fast_tanh(float v) { return __builtin_amdgcn_tanhf(v); }
#else
__device__ __forceinline__ float fast_tanh(float v) { return tanhf(v); }
#endif

__device__ __forceinline__ v16bf combine16(v8bf lo, v8bf hi) {
    return __builtin_shufflevector(lo, hi, 0,1,2,3,4,5,6,7,8,9,10,11,12,13,14,15);
}

// Build one 16x32 bf16 A-fragment from 32 contiguous f32 (two 8-elem runs per lane).
__device__ __forceinline__ v16bf load_x_frag(const float* __restrict__ p0) {
    v16bf a;
#pragma unroll
    for (int e = 0; e < 8; ++e) a[e]     = (bf16_t)p0[e];
#pragma unroll
    for (int e = 0; e < 8; ++e) a[8 + e] = (bf16_t)p0[16 + e];
    return a;
}

// One block = 16 batch rows, runs the entire 512-step recurrence privately.
// 8 waves/block; wave w owns output columns [w*32, w*32+32) (2 N-tiles of 16).
// Wh/Wx bf16 B-fragments live in registers for the whole kernel; h lives in
// ping-pong LDS; one barrier per timestep; no inter-block communication.
__launch_bounds__(256, 1)
__global__ void rnn_wmma_kernel(const float* __restrict__ x,
                                const float* __restrict__ Wx,
                                const float* __restrict__ bx,
                                const float* __restrict__ Wh,
                                const float* __restrict__ bh,
                                float* __restrict__ out)
{
    __shared__ __align__(16) bf16_t hbuf[2][16][HROW];  // ping-pong h state (bf16, padded rows)

    const int tid    = threadIdx.x;
    const int wave   = tid >> 5;
    const int lane   = tid & 31;
    const int hf     = lane >> 4;     // lane half (0/1)
    const int lanelo = lane & 15;
    const int wN     = wave * 32;     // this wave's first output column
    const int bBase  = blockIdx.x * 16;

    // ---------------- one-time setup ----------------
    for (int i = tid; i < 16 * HROW; i += 256)
        ((bf16_t*)hbuf[0])[i] = (bf16_t)0.0f;   // h0 = 0

    // Register-resident Wh B-fragments: B[k][n] = Wh[n][k]  (contiguous rows of Wh)
    v16bf BWh[2][8];
#pragma unroll
    for (int j = 0; j < 2; ++j) {
        const int n = wN + j * 16 + lanelo;
#pragma unroll
        for (int kc = 0; kc < 8; ++kc) {
            const float* wr = Wh + (size_t)n * HID + kc * 32 + hf * 8;
            v16bf f;
#pragma unroll
            for (int e = 0; e < 16; ++e) f[e] = (bf16_t)wr[e + (e >= 8 ? 8 : 0)];
            BWh[j][kc] = f;
        }
    }
    // Register-resident Wx B-fragments
    v16bf BWx[2][2];
#pragma unroll
    for (int j = 0; j < 2; ++j) {
        const int n = wN + j * 16 + lanelo;
#pragma unroll
        for (int kc = 0; kc < 2; ++kc) {
            const float* wr = Wx + (size_t)n * INSZ + kc * 32 + hf * 8;
            v16bf f;
#pragma unroll
            for (int e = 0; e < 16; ++e) f[e] = (bf16_t)wr[e + (e >= 8 ? 8 : 0)];
            BWx[j][kc] = f;
        }
    }

    // Loop-invariant combined bias for this lane's two output columns (registers,
    // so the steady-state loop never touches LDS for bias).
    const int n0 = wN + lanelo;
    const int n1 = wN + 16 + lanelo;
    const float bn0 = bx[n0] + bh[n0];
    const float bn1 = bx[n1] + bh[n1];

    const int m = lanelo;   // A-matrix row this lane supplies (both halves cover M=0..15)

    // Pre-load x A-fragments for t = 0 (software pipeline)
    const float* xr0 = x + ((size_t)(bBase + m)) * INSZ + hf * 8;
    v16bf ax0 = load_x_frag(xr0);
    v16bf ax1 = load_x_frag(xr0 + 32);

    __syncthreads();

    int cur = 0;

    // ---------------- 512-step recurrence ----------------
    for (int t = 0; t < SEQ; ++t) {
        const v8f zero = {0.f, 0.f, 0.f, 0.f, 0.f, 0.f, 0.f, 0.f};

        // Fused input projection with pre-loaded fragments (K = 64)
        v8f acc0 = __builtin_amdgcn_wmma_f32_16x16x32_bf16(false, ax0, false, BWx[0][0],
                                                           (short)0, zero, false, false);
        v8f acc1 = __builtin_amdgcn_wmma_f32_16x16x32_bf16(false, ax0, false, BWx[1][0],
                                                           (short)0, zero, false, false);
        acc0 = __builtin_amdgcn_wmma_f32_16x16x32_bf16(false, ax1, false, BWx[0][1],
                                                       (short)0, acc0, false, false);
        acc1 = __builtin_amdgcn_wmma_f32_16x16x32_bf16(false, ax1, false, BWx[1][1],
                                                       (short)0, acc1, false, false);

        // Prefetch x fragments for t+1: global latency hides behind the
        // 16 recurrent wmmas + epilogue + barrier below.
        if (t < SEQ - 1) {
            const float* xr = x + ((size_t)(t + 1) * BATCH + bBase + m) * INSZ + hf * 8;
            ax0 = load_x_frag(xr);
            ax1 = load_x_frag(xr + 32);
        }

        // Recurrent GEMM: A = h (16 x 256) from LDS, K = 256.
        // Double-buffered A-fragments: chunk kc+1 loads while chunk kc multiplies.
        const bf16_t* hrow = &hbuf[cur][m][hf * 8];
        v16bf a_cur = combine16(*(const v8bf*)(hrow), *(const v8bf*)(hrow + 16));
#pragma unroll
        for (int kc = 0; kc < 8; ++kc) {
            v16bf a_nxt = a_cur;
            if (kc < 7) {
                const bf16_t* hp = hrow + (kc + 1) * 32;
                a_nxt = combine16(*(const v8bf*)(hp), *(const v8bf*)(hp + 16));
            }
            acc0 = __builtin_amdgcn_wmma_f32_16x16x32_bf16(false, a_cur, false, BWh[0][kc],
                                                           (short)0, acc0, false, false);
            acc1 = __builtin_amdgcn_wmma_f32_16x16x32_bf16(false, a_cur, false, BWh[1][kc],
                                                           (short)0, acc1, false, false);
            a_cur = a_nxt;
        }

        // Epilogue: bias + tanh; write f32 h_seq to global, bf16 h to next LDS buffer.
        // C/D layout: lanes 0-15 -> N=lane, VGPR r -> M=r; lanes 16-31 -> M=8+r.
        const int nxt = cur ^ 1;
#pragma unroll
        for (int j = 0; j < 2; ++j) {
            const v8f  accv = j ? acc1 : acc0;
            const int  n    = j ? n1 : n0;
            const float bn  = j ? bn1 : bn0;
#pragma unroll
            for (int r = 0; r < 8; ++r) {
                const int  b = hf * 8 + r;                 // local batch row
                const float v = fast_tanh(accv[r] + bn);
                out[((size_t)t * BATCH + bBase + b) * HID + n] = v;
                hbuf[nxt][b][n] = (bf16_t)v;
                if (t == SEQ - 1)                           // h_last (uniform branch)
                    out[(size_t)SEQ * BATCH * HID + (size_t)(bBase + b) * HID + n] = v;
            }
        }
        __syncthreads();
        cur = nxt;
    }
}

extern "C" void kernel_launch(void* const* d_in, const int* in_sizes, int n_in,
                              void* d_out, int out_size, void* d_ws, size_t ws_size,
                              hipStream_t stream) {
    const float* x  = (const float*)d_in[0];
    const float* Wx = (const float*)d_in[1];
    const float* bx = (const float*)d_in[2];
    const float* Wh = (const float*)d_in[3];
    const float* bh = (const float*)d_in[4];
    float* out = (float*)d_out;

    rnn_wmma_kernel<<<dim3(BATCH / 16), dim3(256), 0, stream>>>(x, Wx, bx, Wh, bh, out);
}
